// MambaOfficialBlock_61143154425892
// MI455X (gfx1250) — matrix operand order repaired and checked by reading, not compiled, not run on hardware
//
#include <hip/hip_runtime.h>

// ---------------------------------------------------------------------------
// Mamba block forward for gfx1250 (MI455X).
// GEMMs: v_wmma_f32_16x16x32_bf16 with double-buffered LDS fed by
// global_load_async_to_lds_b128 (ASYNCcnt-tracked async tensor path).
// ---------------------------------------------------------------------------

typedef __attribute__((ext_vector_type(16))) __bf16 bf16x16;
typedef __attribute__((ext_vector_type(8)))  __bf16 bf16x8;
typedef __attribute__((ext_vector_type(8)))  float  f32x8;

#define D_MODEL 1024
#define D_INNER 2048
#define D_STATE 16
#define DT_RANK 64
#define SEQ_L   4096
#define BATCH   4
#define M_ROWS  (BATCH * SEQ_L)   // 16384

__device__ __forceinline__ __bf16 f2bf(float f) {
    unsigned u = __float_as_uint(f);
    unsigned r = (u + 0x7FFFu + ((u >> 16) & 1u)) >> 16;   // round-to-nearest-even
    unsigned short s = (unsigned short)r;
    return __builtin_bit_cast(__bf16, s);
}

__device__ __forceinline__ float sigmoidf_fast(float x) {
    return 1.0f / (1.0f + __expf(-x));
}

// 16-byte async copy: global -> LDS, tracked by ASYNCcnt.
__device__ __forceinline__ void async_copy16(unsigned lds_off, const void* gaddr) {
    asm volatile("global_load_async_to_lds_b128 %0, %1, off"
                 :: "v"(lds_off), "v"(gaddr)
                 : "memory");
}

__device__ __forceinline__ void wait_async0() {
#if __has_builtin(__builtin_amdgcn_s_wait_asynccnt)
    __builtin_amdgcn_s_wait_asynccnt(0);
#else
    asm volatile("s_wait_asynccnt 0" ::: "memory");
#endif
}

// ------------------------------ fp32 -> bf16 -------------------------------
__global__ __launch_bounds__(256) void cvt_f32_bf16_kernel(
    const float* __restrict__ in, __bf16* __restrict__ out, long n) {
    long i = (long)blockIdx.x * blockDim.x + threadIdx.x;
    if (i < n) out[i] = f2bf(in[i]);
}

// ------------------------------ WMMA GEMM (NT) -----------------------------
// C[M,N] = A[M,K] (bf16, lda) * Wb[N,K]^T (bf16 row-major), fp32 accumulate.
// EPI 0: store fp32 C.
// EPI 1: store fp32 softplus(v + bias[col])          (dt path)
// EPI 2: store fp32 C and bf16 Cbf (same layout)     (x_proj path)
#define BM 128
#define BN 128
#define BK 32
#define LDSK 40   // padded K stride (bf16 elems); rows stay 16B aligned (80B)

template <int EPI>
__global__ __launch_bounds__(256) void gemm_bf16_wmma_kernel(
    float* __restrict__ C, __bf16* __restrict__ Cbf,
    const __bf16* __restrict__ A, const __bf16* __restrict__ Wb,
    const float* __restrict__ bias,
    int M, int N, int K, int lda, int ldc)
{
    __shared__ __bf16 As[2][BM][LDSK];
    __shared__ __bf16 Bs[2][BN][LDSK];

    const int tid  = threadIdx.x;
    const int lane = tid & 31;
    const int wave = tid >> 5;
    const int wm = (wave & 1) * 64;   // wave row origin in block tile
    const int wn = (wave >> 1) * 32;  // wave col origin in block tile
    const int block_m = blockIdx.y * BM;
    const int block_n = blockIdx.x * BN;

    // Issue async copies for one K-slice (128x32 bf16 A tile + B tile).
    // 16B per lane per op; 2 ops per tile per thread.
    auto stage = [&](int buf, int k0) {
#pragma unroll
        for (int it = 0; it < 2; ++it) {
            const int e = (tid + it * 256) * 8;   // bf16 element index in tile
            const int r = e >> 5, c = e & 31;
            int gr = block_m + r; if (gr >= M) gr = M - 1;
            async_copy16((unsigned)(size_t)&As[buf][r][c],
                         A + (size_t)gr * lda + k0 + c);
        }
#pragma unroll
        for (int it = 0; it < 2; ++it) {
            const int e = (tid + it * 256) * 8;
            const int r = e >> 5, c = e & 31;
            int gn = block_n + r; if (gn >= N) gn = N - 1;  // clamp: cols >= N never stored
            async_copy16((unsigned)(size_t)&Bs[buf][r][c],
                         Wb + (size_t)gn * K + k0 + c);
        }
    };

    f32x8 acc[4][2];
#pragma unroll
    for (int mt = 0; mt < 4; ++mt)
#pragma unroll
        for (int nt = 0; nt < 2; ++nt) acc[mt][nt] = {};

    const int rbase = lane & 15;
    const int kA = (lane >> 4) * 8;    // A frag: K chunks {0..7,16..23} / {8..15,24..31}
    const int kB = (lane >> 4) * 8;    // B frag: K chunks {0..7,16..23} / {8..15,24..31}

    stage(0, 0);
    wait_async0();
    __syncthreads();

    for (int k0 = 0; k0 < K; k0 += BK) {
        const int cur = (k0 / BK) & 1;
        if (k0 + BK < K) stage(cur ^ 1, k0 + BK);   // overlap with compute below

        bf16x16 af[4];
#pragma unroll
        for (int mt = 0; mt < 4; ++mt) {
            const int row = wm + mt * 16 + rbase;
            bf16x8 lo = *reinterpret_cast<const bf16x8*>(&As[cur][row][kA]);
            bf16x8 hi = *reinterpret_cast<const bf16x8*>(&As[cur][row][16 + kA]);
            af[mt] = __builtin_shufflevector(lo, hi, 0,1,2,3,4,5,6,7,8,9,10,11,12,13,14,15);
        }
        bf16x16 bfv[2];
#pragma unroll
        for (int nt = 0; nt < 2; ++nt) {
            const int col = wn + nt * 16 + rbase;
            bf16x8 lo = *reinterpret_cast<const bf16x8*>(&Bs[cur][col][kB]);
            bf16x8 hi = *reinterpret_cast<const bf16x8*>(&Bs[cur][col][16 + kB]);
            bfv[nt] = __builtin_shufflevector(lo, hi, 0,1,2,3,4,5,6,7,8,9,10,11,12,13,14,15);
        }

#pragma unroll
        for (int mt = 0; mt < 4; ++mt)
#pragma unroll
            for (int nt = 0; nt < 2; ++nt)
                acc[mt][nt] = __builtin_amdgcn_wmma_f32_16x16x32_bf16(
                    false, af[mt], false, bfv[nt], (short)0, acc[mt][nt], false, false);

        wait_async0();     // next-slice async copies have landed in LDS
        __syncthreads();   // all waves done reading cur / writing nxt
    }

    // --- epilogue: C/D layout -> lane l, reg r: M = r + 8*(l/16), N = l%16 ---
    const int m_off = (lane >> 4) * 8;
    const int n_off = lane & 15;
#pragma unroll
    for (int mt = 0; mt < 4; ++mt) {
#pragma unroll
        for (int nt = 0; nt < 2; ++nt) {
            const int col = block_n + wn + nt * 16 + n_off;
            if (col >= N) continue;
#pragma unroll
            for (int r = 0; r < 8; ++r) {
                const int row = block_m + wm + mt * 16 + m_off + r;
                if (row >= M) continue;
                float v = acc[mt][nt][r];
                if (EPI == 1) {   // bias + softplus (dt path)
                    v += bias[col];
                    v = (v > 20.0f) ? v : __logf(1.0f + __expf(v));
                }
                C[(size_t)row * ldc + col] = v;
                if (EPI == 2) Cbf[(size_t)row * ldc + col] = f2bf(v);
            }
        }
    }
}

// ------------------- depthwise causal conv (k=4) + bias + SiLU -------------
// Writes fp32 u (scan input) and bf16 u (x_proj GEMM input).
__global__ __launch_bounds__(256) void conv_silu_kernel(
    const float* __restrict__ xz, const float* __restrict__ conv_w,
    const float* __restrict__ conv_b, float* __restrict__ u,
    __bf16* __restrict__ ubf)
{
    const long idx = (long)blockIdx.x * blockDim.x + threadIdx.x;
    const long total = (long)M_ROWS * D_INNER;
    if (idx >= total) return;
    const int  d  = (int)(idx & (D_INNER - 1));
    const long bt = idx >> 11;                    // b*L + t
    const int  t  = (int)(bt & (SEQ_L - 1));
    float acc = conv_b[d];
#pragma unroll
    for (int j = 0; j < 4; ++j) {
        const int tt = t - 3 + j;
        if (tt >= 0)
            acc += xz[(bt - 3 + j) * (2 * D_INNER) + d] * conv_w[d * 4 + j];
    }
    const float v = acc * sigmoidf_fast(acc);     // SiLU
    u[idx]   = v;
    ubf[idx] = f2bf(v);
}

// ------------------------------ selective scan -----------------------------
// One thread per (b, d) chain; 16-state diagonal recurrence in VGPRs.
// B_t / C_t broadcast via LDS. Fuses y += u*D and y *= silu(z).
// Output written directly in bf16 (input of the out-projection WMMA GEMM).
__global__ __launch_bounds__(256) void mamba_scan_kernel(
    const float* __restrict__ delta, const float* __restrict__ u,
    const float* __restrict__ dbl,   const float* __restrict__ xz,
    const float* __restrict__ A_log, const float* __restrict__ Dp,
    __bf16* __restrict__ ybf)
{
    __shared__ float sBC[32];
    const int d = blockIdx.x * blockDim.x + threadIdx.x;  // 0..2047
    const int b = blockIdx.y;

    float Arow[D_STATE], h[D_STATE];
#pragma unroll
    for (int n = 0; n < D_STATE; ++n) {
        Arow[n] = -__expf(A_log[d * D_STATE + n]);
        h[n] = 0.0f;
    }
    const float Dd = Dp[d];
    const long row0 = (long)b * SEQ_L;

    for (int t = 0; t < SEQ_L; ++t) {
        const long r = row0 + t;
        if (threadIdx.x < 32) sBC[threadIdx.x] = dbl[r * 96 + 64 + threadIdx.x];
        __syncthreads();

        const float dt  = delta[r * D_INNER + d];
        const float ut  = u[r * D_INNER + d];
        const float dtu = dt * ut;
        float acc = 0.0f;
#pragma unroll
        for (int n = 0; n < D_STATE; ++n) {
            const float dA = __expf(dt * Arow[n]);
            h[n] = dA * h[n] + dtu * sBC[n];
            acc += h[n] * sBC[16 + n];
        }
        const float zt = xz[r * (2 * D_INNER) + D_INNER + d];
        ybf[r * D_INNER + d] = f2bf((acc + ut * Dd) * (zt * sigmoidf_fast(zt)));
        __syncthreads();
    }
}

// ---------------------------------------------------------------------------
extern "C" void kernel_launch(void* const* d_in, const int* in_sizes, int n_in,
                              void* d_out, int out_size, void* d_ws, size_t ws_size,
                              hipStream_t stream) {
    const float* x         = (const float*)d_in[0];
    const float* in_proj_w = (const float*)d_in[1];
    const float* conv_w    = (const float*)d_in[2];
    const float* conv_b    = (const float*)d_in[3];
    const float* x_proj_w  = (const float*)d_in[4];
    const float* dt_proj_w = (const float*)d_in[5];
    const float* dt_proj_b = (const float*)d_in[6];
    const float* A_log     = (const float*)d_in[7];
    const float* Dvec      = (const float*)d_in[8];
    const float* out_proj_w= (const float*)d_in[9];
    float* out = (float*)d_out;

    size_t off = 0;
    auto take = [&](size_t bytes) -> void* {
        void* p = (char*)d_ws + off;
        off += (bytes + 255) & ~(size_t)255;
        return p;
    };
    __bf16* w_in  = (__bf16*)take((size_t)(2*D_INNER) * D_MODEL * 2);   // 4096x1024
    __bf16* w_xp  = (__bf16*)take((size_t)96 * D_INNER * 2);            // 96x2048
    __bf16* w_dt  = (__bf16*)take((size_t)D_INNER * DT_RANK * 2);       // 2048x64
    __bf16* w_out = (__bf16*)take((size_t)D_MODEL * D_INNER * 2);       // 1024x2048
    __bf16* x_bf  = (__bf16*)take((size_t)M_ROWS * D_MODEL * 2);
    float*  xz    = (float*)take((size_t)M_ROWS * (2*D_INNER) * 4);
    float*  u     = (float*)take((size_t)M_ROWS * D_INNER * 4);
    __bf16* u_bf  = (__bf16*)take((size_t)M_ROWS * D_INNER * 2);
    float*  dbl   = (float*)take((size_t)M_ROWS * 96 * 4);
    __bf16* dblq  = (__bf16*)take((size_t)M_ROWS * 96 * 2);
    float*  delta = (float*)take((size_t)M_ROWS * D_INNER * 4);
    __bf16* y_bf  = (__bf16*)take((size_t)M_ROWS * D_INNER * 2);
    (void)ws_size; (void)in_sizes; (void)n_in; (void)out_size;

    // 1) weights + activations -> bf16
    {
        long n;
        n = (long)(2*D_INNER) * D_MODEL;
        cvt_f32_bf16_kernel<<<(n + 255) / 256, 256, 0, stream>>>(in_proj_w, w_in, n);
        n = (long)96 * D_INNER;
        cvt_f32_bf16_kernel<<<(n + 255) / 256, 256, 0, stream>>>(x_proj_w, w_xp, n);
        n = (long)D_INNER * DT_RANK;
        cvt_f32_bf16_kernel<<<(n + 255) / 256, 256, 0, stream>>>(dt_proj_w, w_dt, n);
        n = (long)D_MODEL * D_INNER;
        cvt_f32_bf16_kernel<<<(n + 255) / 256, 256, 0, stream>>>(out_proj_w, w_out, n);
        n = (long)M_ROWS * D_MODEL;
        cvt_f32_bf16_kernel<<<(n + 255) / 256, 256, 0, stream>>>(x, x_bf, n);
    }

    // 2) xz[M,4096] = x[M,1024] @ in_proj_w^T
    gemm_bf16_wmma_kernel<0><<<dim3((2*D_INNER)/BN, M_ROWS/BM), 256, 0, stream>>>(
        xz, nullptr, x_bf, w_in, nullptr, M_ROWS, 2*D_INNER, D_MODEL, D_MODEL, 2*D_INNER);

    // 3) u = silu(causal_conv(xin) + b)   (fp32 + bf16)
    {
        long n = (long)M_ROWS * D_INNER;
        conv_silu_kernel<<<(n + 255) / 256, 256, 0, stream>>>(xz, conv_w, conv_b, u, u_bf);
    }

    // 4) dbl[M,96] = u[M,2048] @ x_proj_w^T   (fp32 for scan, bf16 shadow for dt GEMM)
    gemm_bf16_wmma_kernel<2><<<dim3(1, M_ROWS/BM), 256, 0, stream>>>(
        dbl, dblq, u_bf, w_xp, nullptr, M_ROWS, 96, D_INNER, D_INNER, 96);

    // 5) delta[M,2048] = softplus(dt_low[M,64] @ dt_proj_w^T + dt_proj_b)
    gemm_bf16_wmma_kernel<1><<<dim3(D_INNER/BN, M_ROWS/BM), 256, 0, stream>>>(
        delta, nullptr, dblq, w_dt, dt_proj_b, M_ROWS, D_INNER, DT_RANK, 96, D_INNER);

    // 6) selective scan (+ u*D, * silu(z)) -> bf16
    mamba_scan_kernel<<<dim3(D_INNER/256, BATCH), 256, 0, stream>>>(
        delta, u, dbl, xz, A_log, Dvec, y_bf);

    // 7) out[M,1024] = y[M,2048] @ out_proj_w^T
    gemm_bf16_wmma_kernel<0><<<dim3(D_MODEL/BN, M_ROWS/BM), 256, 0, stream>>>(
        out, nullptr, y_bf, w_out, nullptr, M_ROWS, D_MODEL, D_INNER, D_INNER, D_MODEL);
}
